// AttentionModule_53463752901436
// MI455X (gfx1250) — compile-verified, matrix operand order checked by reading, not run on hardware
//
#include <hip/hip_runtime.h>
#include <math.h>

typedef __attribute__((ext_vector_type(16))) __bf16 v16bf;
typedef __attribute__((ext_vector_type(8)))  __bf16 v8bf;
typedef __attribute__((ext_vector_type(8)))  float  v8f;

static constexpr int Bc = 2;
static constexpr int Sc = 2048;
static constexpr int Dc = 2048;
static constexpr int Hc = 16;
static constexpr int DHc = 128;
static constexpr int Mc = Bc * Sc;     // 4096 rows for the projection GEMMs

// ---------------- helpers ----------------

__device__ __forceinline__ __bf16 to_bf16(float f) {
  unsigned u = __builtin_bit_cast(unsigned, f);
  unsigned r = u + 0x7FFFu + ((u >> 16) & 1u);   // round-to-nearest-even
  unsigned short h = (unsigned short)(r >> 16);
  return __builtin_bit_cast(__bf16, h);
}

// LDS byte offset of a __shared__ element: generic LDS addr = {SHARED_BASE, off[31:0]}
__device__ __forceinline__ unsigned lds_off(const void* p) {
  return (unsigned)(uintptr_t)p;
}

// CDNA5 async DMA: global -> LDS, 16B per lane, GVS addressing, ASYNCcnt-tracked.
__device__ __forceinline__ void async_load_b128(unsigned ldsoff, unsigned voff,
                                                const void* sbase) {
  asm volatile("global_load_async_to_lds_b128 %0, %1, %2"
               :: "v"(ldsoff), "v"(voff), "s"(sbase) : "memory");
}

__device__ __forceinline__ void wait_async() {
  asm volatile("s_wait_asynccnt 0" ::: "memory");
}

// Load one 16x32 bf16 WMMA A/B fragment. Per CDNA5 ISA (CDNA5 ISA):
// lane L holds row (L&15); first 8 elems are K = 8*(L>>4)+0..7, next 8 are
// K = 16 + 8*(L>>4)+0..7  -> two 16-byte chunks, 32 bytes apart.
__device__ __forceinline__ v16bf ld_frag(const __bf16* base, int ldHalf, int rowBase) {
  const int lane = threadIdx.x & 31;
  const __bf16* p = base + (size_t)(rowBase + (lane & 15)) * ldHalf + ((lane >> 4) & 1) * 8;
  v8bf lo = *(const v8bf*)p;
  v8bf hi = *(const v8bf*)(p + 16);
  return __builtin_shufflevector(lo, hi, 0,1,2,3,4,5,6,7,8,9,10,11,12,13,14,15);
}

__device__ __forceinline__ v8f wmma_bf16(v16bf a, v16bf b, v8f c) {
  return __builtin_amdgcn_wmma_f32_16x16x32_bf16(false, a, false, b, (short)0, c, false, false);
}

// ---------------- elementwise kernels ----------------

__global__ __launch_bounds__(256) void cvt_f32_bf16(const float* __restrict__ src,
                                                    unsigned short* __restrict__ dstraw, int n) {
  __bf16* dst = (__bf16*)dstraw;
  int i = blockIdx.x * blockDim.x + threadIdx.x;
  if (i < n) dst[i] = to_bf16(src[i]);
}

// RoPE + pack [B,S,H,DH] fp32 -> [B,H,S,DH] bf16 (Q gets 1/sqrt(DH) folded in)
__global__ __launch_bounds__(256) void rope_pack(const float* __restrict__ src,
                                                 const int* __restrict__ pos_ids,
                                                 unsigned short* __restrict__ dstraw,
                                                 float scale) {
  __bf16* dst = (__bf16*)dstraw;
  int idx = blockIdx.x * blockDim.x + threadIdx.x;   // B*S*H*64 = 2^22 threads
  if (idx >= Bc * Sc * Hc * 64) return;
  int j = idx & 63;
  int h = (idx >> 6) & (Hc - 1);
  int s = (idx >> 10) & (Sc - 1);
  int b = idx >> 21;
  int m = b * Sc + s;
  float pos = (float)pos_ids[m];
  float inv = __powf(10000.0f, -(float)(2 * j) / (float)DHc);
  float f = pos * inv;
  float sn, cs;
  sincosf(f, &sn, &cs);
  float x1 = src[(size_t)m * Dc + h * DHc + j];
  float x2 = src[(size_t)m * Dc + h * DHc + j + 64];
  size_t o = ((size_t)(b * Hc + h) * Sc + s) * DHc + j;
  dst[o]      = to_bf16((x1 * cs - x2 * sn) * scale);
  dst[o + 64] = to_bf16((x1 * sn + x2 * cs) * scale);
}

// Pack V transposed: [B,S,H,DH] fp32 -> [B,H,DH,S] bf16
__global__ __launch_bounds__(256) void pack_vt(const float* __restrict__ src,
                                               unsigned short* __restrict__ dstraw) {
  __bf16* dst = (__bf16*)dstraw;
  int idx = blockIdx.x * blockDim.x + threadIdx.x;   // 2^23 threads
  if (idx >= Bc * Sc * Hc * DHc) return;
  int d = idx & (DHc - 1);
  int h = (idx >> 7) & (Hc - 1);
  int s = (idx >> 11) & (Sc - 1);
  int b = idx >> 22;
  dst[((size_t)(b * Hc + h) * DHc + d) * Sc + s] =
      to_bf16(src[((size_t)(b * Sc + s)) * Dc + h * DHc + d]);
}

// ---------------- bf16 NT GEMM: C[4096,2048] = A[4096,2048] * Bw[2048,2048]^T ----
// block: 128x128 tile, 8 waves as 4(M)x2(N), each wave 32x64 (2x4 WMMA tiles)
// Dims are compile-time so all store offsets become immediates.

__global__ __launch_bounds__(256) void gemm_bf16_nt(const unsigned short* __restrict__ Araw,
                                                    const unsigned short* __restrict__ Braw,
                                                    float* __restrict__ C) {
  constexpr int N = Dc, K = Dc;
  const __bf16* A  = (const __bf16*)Araw;
  const __bf16* Bw = (const __bf16*)Braw;
  __shared__ __align__(16) __bf16 At[128 * 40];   // 32 K-cols + 8 pad -> 80B row stride
  __shared__ __align__(16) __bf16 Bt[128 * 40];
  const int t = threadIdx.x;
  const int lane = t & 31, wid = t >> 5;
  const int half = (lane >> 4) & 1, l16 = lane & 15;
  const int m0 = blockIdx.y * 128, n0 = blockIdx.x * 128;
  const int wm = (wid & 3) * 32, wn = (wid >> 2) * 64;
  // per-thread staging coordinates (512 16B chunks per tile, 2 per thread)
  const int r0 = t >> 2, c0 = (t & 3) * 8;
  const int r1 = (t + 256) >> 2, c1 = ((t + 256) & 3) * 8;
  v8f acc[2][4] = {};
  for (int kk = 0; kk < K; kk += 32) {
    __syncthreads();
    async_load_b128(lds_off(&At[r0 * 40 + c0]),
                    (unsigned)(((m0 + r0) * K + kk + c0) * 2), A);
    async_load_b128(lds_off(&At[r1 * 40 + c1]),
                    (unsigned)(((m0 + r1) * K + kk + c1) * 2), A);
    async_load_b128(lds_off(&Bt[r0 * 40 + c0]),
                    (unsigned)(((n0 + r0) * K + kk + c0) * 2), Bw);
    async_load_b128(lds_off(&Bt[r1 * 40 + c1]),
                    (unsigned)(((n0 + r1) * K + kk + c1) * 2), Bw);
    wait_async();
    __syncthreads();
    v16bf a[2], b[4];
    #pragma unroll
    for (int i = 0; i < 2; ++i) a[i] = ld_frag(At, 40, wm + i * 16);
    #pragma unroll
    for (int j = 0; j < 4; ++j) b[j] = ld_frag(Bt, 40, wn + j * 16);
    #pragma unroll
    for (int i = 0; i < 2; ++i)
      #pragma unroll
      for (int j = 0; j < 4; ++j)
        acc[i][j] = wmma_bf16(a[i], b[j], acc[i][j]);
  }
  // C-fragment layout: VGPR r -> row r (lanes 0-15) / r+8 (lanes 16-31), col = lane&15
  float* cbase = C + (size_t)(m0 + wm + half * 8) * N + n0 + wn + l16;
  #pragma unroll
  for (int i = 0; i < 2; ++i)
    #pragma unroll
    for (int j = 0; j < 4; ++j)
      #pragma unroll
      for (int r = 0; r < 8; ++r)
        cbase[(i * 16 + r) * N + j * 16] = acc[i][j][r];
}

// ---------------- flash attention ----------------
// grid: (S/128, H, B); block 256 = 8 waves; wave owns 16 query rows.

__global__ __launch_bounds__(256) void flash_attn(const unsigned short* __restrict__ Qraw,
                                                  const unsigned short* __restrict__ Kraw,
                                                  const unsigned short* __restrict__ Vraw,
                                                  const float* __restrict__ mask,
                                                  unsigned short* __restrict__ AOraw) {
  const __bf16* Qb  = (const __bf16*)Qraw;
  const __bf16* Kb  = (const __bf16*)Kraw;
  const __bf16* Vtb = (const __bf16*)Vraw;
  __bf16* AOb = (__bf16*)AOraw;

  __shared__ __align__(16) __bf16 Kt[32 * 136];     // 32 keys x 128 dh (+16B pad/row)
  __shared__ __align__(16) __bf16 Vt[128 * 40];     // 128 dh x 32 keys (+16B pad/row)
  __shared__ __align__(16) __bf16 Pt[8 * 16 * 40];  // per-wave P transpose buffer

  const int t = threadIdx.x, lane = t & 31, wid = t >> 5;
  const int l16 = lane & 15, half = (lane >> 4) & 1;
  const int b = blockIdx.z, h = blockIdx.y;
  const int qm0 = blockIdx.x * 128 + wid * 16;

  const __bf16* qbase = Qb + (size_t)(b * Hc + h) * Sc * DHc;
  const __bf16* kbase = Kb + (size_t)(b * Hc + h) * Sc * DHc;
  const __bf16* vbase = Vtb + (size_t)(b * Hc + h) * DHc * Sc;

  v16bf qf[4];
  #pragma unroll
  for (int i = 0; i < 4; ++i) qf[i] = ld_frag(qbase + i * 32, DHc, qm0);

  float mstate[8], lstate[8];
  v8f o[8] = {};
  #pragma unroll
  for (int r = 0; r < 8; ++r) { mstate[r] = -1e30f; lstate[r] = 0.0f; }
  __bf16* pbuf = &Pt[wid * 16 * 40];

  // per-thread staging coordinates
  const int kr0 = t >> 4, kc0 = (t & 15) * 8;           // K tile: 16 chunks/row
  const int kr1 = (t + 256) >> 4, kc1 = ((t + 256) & 15) * 8;
  const int vr0 = t >> 2, vc0 = (t & 3) * 8;            // V tile: 4 chunks/row
  const int vr1 = (t + 256) >> 2, vc1 = ((t + 256) & 3) * 8;

  for (int kt = 0; kt < Sc / 32; ++kt) {
    __syncthreads();                               // previous tile reads done
    if (kt + 1 < Sc / 32)
      __builtin_prefetch(&kbase[(size_t)((kt + 1) * 32) * DHc], 0, 3);
    async_load_b128(lds_off(&Kt[kr0 * 136 + kc0]),
                    (unsigned)(((kt * 32 + kr0) * DHc + kc0) * 2), kbase);
    async_load_b128(lds_off(&Kt[kr1 * 136 + kc1]),
                    (unsigned)(((kt * 32 + kr1) * DHc + kc1) * 2), kbase);
    async_load_b128(lds_off(&Vt[vr0 * 40 + vc0]),
                    (unsigned)((vr0 * Sc + kt * 32 + vc0) * 2), vbase);
    async_load_b128(lds_off(&Vt[vr1 * 40 + vc1]),
                    (unsigned)((vr1 * Sc + kt * 32 + vc1) * 2), vbase);
    wait_async();
    __syncthreads();

    // scores: 16 queries x 32 keys, K-dim = DH = 128 -> 4 WMMAs per 16-col tile
    v8f sc0 = {}, sc1 = {};
    #pragma unroll
    for (int kk = 0; kk < 4; ++kk) {
      v16bf k0 = ld_frag(&Kt[kk * 32], 136, 0);
      v16bf k1 = ld_frag(&Kt[kk * 32], 136, 16);
      sc0 = wmma_bf16(qf[kk], k0, sc0);
      sc1 = wmma_bf16(qf[kk], k1, sc1);
    }

    // online softmax (Q was pre-scaled by 1/sqrt(DH); add mask here)
    #pragma unroll
    for (int r = 0; r < 8; ++r) {
      int qrow = qm0 + half * 8 + r;
      float s0 = sc0[r] + mask[(size_t)qrow * Sc + kt * 32 + l16];
      float s1 = sc1[r] + mask[(size_t)qrow * Sc + kt * 32 + 16 + l16];
      float v = fmaxf(s0, s1);
      #pragma unroll
      for (int i = 8; i > 0; i >>= 1) v = fmaxf(v, __shfl_xor(v, i, 16));
      float mn = fmaxf(mstate[r], v);
      float alpha = __expf(mstate[r] - mn);
      float p0 = __expf(s0 - mn), p1 = __expf(s1 - mn);
      float rs = p0 + p1;
      #pragma unroll
      for (int i = 8; i > 0; i >>= 1) rs += __shfl_xor(rs, i, 16);
      lstate[r] = lstate[r] * alpha + rs;
      mstate[r] = mn;
      #pragma unroll
      for (int tt = 0; tt < 8; ++tt) o[tt][r] *= alpha;
      pbuf[(half * 8 + r) * 40 + l16]      = to_bf16(p0);   // C-layout -> row-major
      pbuf[(half * 8 + r) * 40 + 16 + l16] = to_bf16(p1);
    }
    __syncthreads();                               // P visible, Vt still valid

    v16bf pf = ld_frag(pbuf, 40, 0);               // reload P in A-fragment layout
    #pragma unroll
    for (int tt = 0; tt < 8; ++tt) {
      v16bf vf = ld_frag(Vt, 40, tt * 16);
      o[tt] = wmma_bf16(pf, vf, o[tt]);
    }
  }

  // normalize + store bf16 attn-out [B*S, H*DH]
  __bf16* aob = AOb + (size_t)(b * Sc + qm0 + half * 8) * (Hc * DHc) + h * DHc + l16;
  #pragma unroll
  for (int r = 0; r < 8; ++r) {
    float inv = 1.0f / lstate[r];
    #pragma unroll
    for (int tt = 0; tt < 8; ++tt)
      aob[(size_t)r * (Hc * DHc) + tt * 16] = to_bf16(o[tt][r] * inv);
  }
}

// ---------------- launch ----------------

extern "C" void kernel_launch(void* const* d_in, const int* in_sizes, int n_in,
                              void* d_out, int out_size, void* d_ws, size_t ws_size,
                              hipStream_t stream) {
  const float* x    = (const float*)d_in[0];
  const float* mask = (const float*)d_in[1];
  const int*   pos  = (const int*)d_in[2];
  const float* wq   = (const float*)d_in[3];
  const float* wk   = (const float*)d_in[4];
  const float* wv   = (const float*)d_in[5];
  const float* wo   = (const float*)d_in[6];
  float* out = (float*)d_out;

  char* p = (char*)d_ws;
  auto alloc = [&](size_t bytes) {
    char* r = p; p += (bytes + 255) & ~(size_t)255; return r;
  };
  unsigned short* xb  = (unsigned short*)alloc((size_t)Mc * Dc * 2);
  unsigned short* wqb = (unsigned short*)alloc((size_t)Dc * Dc * 2);
  unsigned short* wkb = (unsigned short*)alloc((size_t)Dc * Dc * 2);
  unsigned short* wvb = (unsigned short*)alloc((size_t)Dc * Dc * 2);
  unsigned short* wob = (unsigned short*)alloc((size_t)Dc * Dc * 2);
  unsigned short* Qb  = (unsigned short*)alloc((size_t)Mc * Dc * 2);
  unsigned short* Kbf = (unsigned short*)alloc((size_t)Mc * Dc * 2);
  unsigned short* Vtb = (unsigned short*)alloc((size_t)Mc * Dc * 2);
  unsigned short* AOb = (unsigned short*)alloc((size_t)Mc * Dc * 2);
  float*          Pf  = (float*)alloc((size_t)Mc * Dc * 4);
  (void)in_sizes; (void)n_in; (void)out_size; (void)ws_size;

  const int nX = Mc * Dc, nW = Dc * Dc;
  cvt_f32_bf16<<<(nX + 255) / 256, 256, 0, stream>>>(x, xb, nX);
  cvt_f32_bf16<<<(nW + 255) / 256, 256, 0, stream>>>(wq, wqb, nW);
  cvt_f32_bf16<<<(nW + 255) / 256, 256, 0, stream>>>(wk, wkb, nW);
  cvt_f32_bf16<<<(nW + 255) / 256, 256, 0, stream>>>(wv, wvb, nW);
  cvt_f32_bf16<<<(nW + 255) / 256, 256, 0, stream>>>(wo, wob, nW);

  dim3 ggrid(Dc / 128, Mc / 128);
  const float qscale = 0.08838834764831845f;   // 1/sqrt(128)
  const int nRope = Bc * Sc * Hc * 64, nPV = Bc * Sc * Hc * DHc;

  gemm_bf16_nt<<<ggrid, 256, 0, stream>>>(xb, wqb, Pf);
  rope_pack<<<(nRope + 255) / 256, 256, 0, stream>>>(Pf, pos, Qb, qscale);

  gemm_bf16_nt<<<ggrid, 256, 0, stream>>>(xb, wkb, Pf);
  rope_pack<<<(nRope + 255) / 256, 256, 0, stream>>>(Pf, pos, Kbf, 1.0f);

  gemm_bf16_nt<<<ggrid, 256, 0, stream>>>(xb, wvb, Pf);
  pack_vt<<<(nPV + 255) / 256, 256, 0, stream>>>(Pf, Vtb);

  flash_attn<<<dim3(Sc / 128, Hc, Bc), 256, 0, stream>>>(Qb, Kbf, Vtb, mask, AOb);

  gemm_bf16_nt<<<ggrid, 256, 0, stream>>>(AOb, wob, out);
}
